// GroupMambaBlock_25898652795558
// MI455X (gfx1250) — matrix operand order repaired and checked by reading, not compiled, not run on hardware
//
#include <hip/hip_runtime.h>

typedef unsigned short u16;
typedef unsigned int   u32;
typedef __attribute__((ext_vector_type(16))) __bf16 v16bf;
typedef __attribute__((ext_vector_type(8)))  float  v8f;

constexpr int Bz = 4, Cz = 256, Lz = 2048, Gz = 4, Dz = 64, Ez = 128, Nz = 16, Rz = 4, Kz = 4;
constexpr int F2 = 256;            // 2*E
constexpr int XPR = Rz + 2 * Nz;   // 36
constexpr int Sz = 16;             // scan chunks
constexpr int CHL = Lz / Sz;       // 128 steps per chunk

// ---------- helpers ----------
__device__ __forceinline__ u16 f2bf(float f) {
  union { float f; u32 u; } v; v.f = f;
  u32 r = v.u + 0x7FFFu + ((v.u >> 16) & 1u);   // round-to-nearest-even
  return (u16)(r >> 16);
}
__device__ __forceinline__ u32 pack2(float a, float b) {
  return (u32)f2bf(a) | ((u32)f2bf(b) << 16);
}
__device__ __forceinline__ float sigm(float x) { return 1.f / (1.f + __expf(-x)); }
__device__ __forceinline__ float softplusf(float x) { return x > 20.f ? x : log1pf(__expf(x)); }

union Frag { v16bf v; uint4 q[2]; };
__device__ __forceinline__ v8f vzero() { v8f v = {0.f,0.f,0.f,0.f,0.f,0.f,0.f,0.f}; return v; }

// One wave computes a 16(M) x 64(N) tile: 4 accumulators, K in steps of 32.
__device__ __forceinline__ void wmma_16x64(const u16* __restrict__ A, int lda,
                                           const u16* __restrict__ Bp, int ldb,
                                           int K, v8f acc[4]) {
  const int lane = threadIdx.x & 31;
  const int m  = lane & 15;
  const int kb = (lane & 16) ? 8 : 0;           // A-matrix half-wave K offset
  for (int k0 = 0; k0 < K; k0 += 32) {
    Frag a;
    const u16* ap = A + (size_t)m * lda + k0 + kb;
    a.q[0] = *(const uint4*)(ap);               // K = base .. base+7
    a.q[1] = *(const uint4*)(ap + 16);          // K = base+16 .. base+23
    const u16* bp = Bp + (size_t)(k0 + kb) * ldb + m;   // B column n = lane%16, K = base+s (+8 for s>=8)
#pragma unroll
    for (int nt = 0; nt < 4; ++nt) {
      Frag b;
      u16 bt[16];
#pragma unroll
      for (int s = 0; s < 16; ++s) bt[s] = bp[(size_t)(s + (s & 8)) * ldb + nt * 16];
      __builtin_memcpy(&b.v, bt, sizeof(bt));
      acc[nt] = __builtin_amdgcn_wmma_f32_16x16x32_bf16(
          false, a.v, false, b.v, (short)0, acc[nt], false, false);
    }
  }
}

// ---------- K1: channel LN (+ group LN, + partial sums for s) ----------
__global__ void k_ln(const float* __restrict__ x, const float* __restrict__ ln_w,
                     const float* __restrict__ ln_b, const float* __restrict__ glw,
                     const float* __restrict__ glb, float* __restrict__ xn,
                     u16* __restrict__ hg, float* __restrict__ s_part) {
  __shared__ float tile[Cz][33];
  __shared__ float part8[8][Cz];
  const int tid = threadIdx.x;
  const int b  = blockIdx.x >> 6;          // Lz/32 = 64 chunks per batch
  const int l0 = (blockIdx.x & 63) * 32;

  const int lp0 = tid & 31;
  for (int c = tid >> 5; c < Cz; c += 8)
    tile[c][lp0] = x[((size_t)(b * Cz + c)) * Lz + l0 + lp0];
  __syncthreads();

  const int wave = tid >> 5, lane = tid & 31;
  const int g = lane >> 3;                 // 8 lanes (64 channels) per group
  float wsum[8];
#pragma unroll
  for (int j = 0; j < 8; ++j) wsum[j] = 0.f;

  for (int lp = wave; lp < 32; lp += 8) {
    const int l = l0 + lp;
    float v[8], s1 = 0.f, s2 = 0.f;
#pragma unroll
    for (int j = 0; j < 8; ++j) { float t = tile[lane * 8 + j][lp]; v[j] = t; s1 += t; s2 += t * t; }
#pragma unroll
    for (int off = 16; off >= 1; off >>= 1) { s1 += __shfl_xor(s1, off, 32); s2 += __shfl_xor(s2, off, 32); }
    const float mean = s1 * (1.f / Cz);
    const float rstd = rsqrtf(s2 * (1.f / Cz) - mean * mean + 1e-5f);

    float xv[8], g1 = 0.f, g2 = 0.f;
#pragma unroll
    for (int j = 0; j < 8; ++j) {
      const int c = lane * 8 + j;
      float t = (v[j] - mean) * rstd * ln_w[c] + ln_b[c];
      xv[j] = t; g1 += t; g2 += t * t; wsum[j] += t;
    }
    float4* xr = (float4*)(xn + ((size_t)(b * Lz + l)) * Cz + lane * 8);
    xr[0] = make_float4(xv[0], xv[1], xv[2], xv[3]);
    xr[1] = make_float4(xv[4], xv[5], xv[6], xv[7]);

    // group LN over D=64 (8 lanes x 8 values)
#pragma unroll
    for (int off = 4; off >= 1; off >>= 1) { g1 += __shfl_xor(g1, off, 32); g2 += __shfl_xor(g2, off, 32); }
    const float gm  = g1 * (1.f / Dz);
    const float grs = rsqrtf(g2 * (1.f / Dz) - gm * gm + 1e-5f);
    u32 hp[4];
#pragma unroll
    for (int j = 0; j < 8; j += 2) {
      const int d = (lane * 8 + j) & 63;
      float a0 = (xv[j]   - gm) * grs * glw[g * Dz + d]     + glb[g * Dz + d];
      float a1 = (xv[j+1] - gm) * grs * glw[g * Dz + d + 1] + glb[g * Dz + d + 1];
      hp[j >> 1] = pack2(a0, a1);
    }
    *(uint4*)(hg + ((size_t)(b * Gz + g) * Lz + l) * Dz + (lane & 7) * 8) =
        make_uint4(hp[0], hp[1], hp[2], hp[3]);
  }
#pragma unroll
  for (int j = 0; j < 8; ++j) part8[wave][lane * 8 + j] = wsum[j];
  __syncthreads();
  float t = 0.f;
#pragma unroll
  for (int w = 0; w < 8; ++w) t += part8[w][tid];
  s_part[(size_t)blockIdx.x * Cz + tid] = t;     // deterministic per-block partial
}

// ---------- K2: channel-attention gate ----------
__global__ void k_gate(const float* __restrict__ s_part, const float* __restrict__ w1,
                       const float* __restrict__ b1, const float* __restrict__ w2,
                       const float* __restrict__ b2, float* __restrict__ wgate) {
  __shared__ float sv[Cz];
  __shared__ float hv[Cz / 4];
  const int b = blockIdx.x, tid = threadIdx.x;
  float acc = 0.f;
  for (int ch = 0; ch < 64; ++ch) acc += s_part[(size_t)(b * 64 + ch) * Cz + tid];
  sv[tid] = acc * (1.f / Lz);
  __syncthreads();
  if (tid < Cz / 4) {
    float a = b1[tid];
    for (int c = 0; c < Cz; ++c) a += sv[c] * w1[tid * Cz + c];
    hv[tid] = a > 0.f ? a : 0.f;
  }
  __syncthreads();
  float a = b2[tid];
  for (int j = 0; j < Cz / 4; ++j) a += hv[j] * w2[tid * (Cz / 4) + j];
  wgate[b * Cz + tid] = sigm(a);
}

// ---------- K3: transpose+convert weights to bf16 once per launch ----------
__global__ void k_convert(const float* __restrict__ g_in_w, const float* __restrict__ g_out_w,
                          const float* __restrict__ proj_w, u16* __restrict__ Wt,
                          u16* __restrict__ Wot, u16* __restrict__ Wp) {
  const int i = blockIdx.x * blockDim.x + threadIdx.x;
  const int n1 = Gz * Dz * F2, n2 = Gz * Ez * Dz, n3 = Cz * Cz;
  if (i < n1) {                                   // Wt[g][d][f] = g_in_w[g][f][d]
    int g = i / (Dz * F2), rem = i % (Dz * F2), d = rem / F2, f = rem % F2;
    Wt[i] = f2bf(g_in_w[((size_t)g * F2 + f) * Dz + d]);
  } else if (i < n1 + n2) {                       // Wot[g][e][d] = g_out_w[g][d][e]
    int j = i - n1;
    int g = j / (Ez * Dz), rem = j % (Ez * Dz), e = rem / Dz, d = rem % Dz;
    Wot[j] = f2bf(g_out_w[((size_t)g * Dz + d) * Ez + e]);
  } else if (i < n1 + n2 + n3) {                  // Wp[ci][co] = proj_w[co][ci]
    int j = i - n1 - n2, ci = j / Cz, co = j % Cz;
    Wp[j] = f2bf(proj_w[(size_t)co * Cz + ci]);
  }
}

// ---------- K4: in_proj GEMM  xz = hg @ Win^T  (per b,g: 2048x64 @ 64x256) ----------
__global__ void k_gemm_in(const u16* __restrict__ hg, const u16* __restrict__ Wt,
                          float* __restrict__ xz) {
  const int wave = blockIdx.x * (blockDim.x >> 5) + (threadIdx.x >> 5);
  const int n64 = wave & 3;
  const int mt  = (wave >> 2) & 127;
  const int bg  = wave >> 9;
  const int g   = bg & 3;
  const u16* A  = hg + ((size_t)bg * Lz + mt * 16) * Dz;
  const u16* Bp = Wt + (size_t)g * Dz * F2 + n64 * 64;
  v8f acc[4];
#pragma unroll
  for (int i = 0; i < 4; ++i) acc[i] = vzero();
  wmma_16x64(A, Dz, Bp, F2, Dz, acc);
  const int lane = threadIdx.x & 31, n = lane & 15, hi = lane >> 4;
  float* dst = xz + ((size_t)bg * Lz + mt * 16 + hi * 8) * F2 + n64 * 64 + n;
#pragma unroll
  for (int nt = 0; nt < 4; ++nt)
#pragma unroll
    for (int r = 0; r < 8; ++r) dst[(size_t)r * F2 + nt * 16] = acc[nt][r];
}

// ---------- K5: causal conv + SiLU + x_dbl + dt (fused, LDS-cached weights) ----------
__global__ void k_conv(const float* __restrict__ xz, const float* __restrict__ g_conv_w,
                       const float* __restrict__ g_conv_b, const float* __restrict__ g_xproj_w,
                       const float* __restrict__ g_dt_w, const float* __restrict__ g_dt_b,
                       float* __restrict__ uc, float* __restrict__ dt,
                       float* __restrict__ Bm, float* __restrict__ Cm) {
  __shared__ float xw_s[XPR * Ez];
  __shared__ float cw_s[Ez * Kz];
  __shared__ float cb_s[Ez];
  __shared__ float dtw_s[Ez * Rz];
  __shared__ float dtb_s[Ez];
  const int tid = threadIdx.x;
  const int chunk = blockIdx.x & 63;
  const int bg = blockIdx.x >> 6;
  const int g  = bg & 3;
  const int l0 = chunk * 32;
  for (int i = tid; i < XPR * Ez; i += 256) xw_s[i]  = g_xproj_w[(size_t)g * XPR * Ez + i];
  for (int i = tid; i < Ez * Kz;  i += 256) cw_s[i]  = g_conv_w[(size_t)g * Ez * Kz + i];
  for (int i = tid; i < Ez * Rz;  i += 256) dtw_s[i] = g_dt_w[(size_t)g * Ez * Rz + i];
  if (tid < Ez) { cb_s[tid] = g_conv_b[g * Ez + tid]; dtb_s[tid] = g_dt_b[g * Ez + tid]; }
  __syncthreads();

  const int wave = tid >> 5, lane = tid & 31;
  for (int li = 0; li < 4; ++li) {
    const int l = l0 + wave * 4 + li;
    float ucv[4];
#pragma unroll
    for (int j = 0; j < 4; ++j) {
      const int e = lane * 4 + j;
      float a = cb_s[e];
#pragma unroll
      for (int k = 0; k < Kz; ++k) {
        const int ls = l - (Kz - 1) + k;
        const float uv = (ls >= 0) ? xz[((size_t)bg * Lz + ls) * F2 + e] : 0.f;
        a += uv * cw_s[e * Kz + k];
      }
      const float s = a * sigm(a);                         // SiLU
      ucv[j] = s;
      uc[((size_t)bg * Lz + l) * Ez + e] = s;
    }
    float dl[4]; float bmv = 0.f, cmv = 0.f;
#pragma unroll
    for (int r = 0; r < XPR; ++r) {
      float p = 0.f;
#pragma unroll
      for (int j = 0; j < 4; ++j) p += ucv[j] * xw_s[r * Ez + lane * 4 + j];
#pragma unroll
      for (int off = 16; off >= 1; off >>= 1) p += __shfl_xor(p, off, 32);
      if (r < Rz) dl[r] = p;
      else if (r < Rz + Nz) { if (lane == r - Rz) bmv = p; }
      else                  { if (lane == r - Rz - Nz) cmv = p; }
    }
    if (lane < Nz) {
      const size_t bi = ((size_t)bg * Lz + l) * Nz + lane;
      Bm[bi] = bmv; Cm[bi] = cmv;
    }
#pragma unroll
    for (int j = 0; j < 4; ++j) {
      const int e = lane * 4 + j;
      float a = dtb_s[e];
#pragma unroll
      for (int r = 0; r < Rz; ++r) a += dl[r] * dtw_s[e * Rz + r];
      dt[((size_t)bg * Lz + l) * Ez + e] = softplusf(a);
    }
  }
}

// ---------- K6a: chunked scan pass A: per-chunk local state q and sum(dt) ----------
// Chunk transfer: h_out = exp(A * sum(dt)) * h_in + q   (since prod exp(dt_t*A) = exp(A*sum dt))
__global__ void k_scan_a(const float* __restrict__ dt, const float* __restrict__ uc,
                         const float* __restrict__ Bm, const float* __restrict__ Alog,
                         float* __restrict__ sdt_buf, float* __restrict__ q_buf) {
  const int bg = blockIdx.x >> 4;
  const int s  = blockIdx.x & 15;
  const int g = bg & 3, e = threadIdx.x;
  float A[Nz];
#pragma unroll
  for (int n = 0; n < Nz; ++n) A[n] = -__expf(Alog[((size_t)g * Ez + e) * Nz + n]);
  float h[Nz];
#pragma unroll
  for (int n = 0; n < Nz; ++n) h[n] = 0.f;
  float sdt = 0.f;
  __shared__ float Bs[16][Nz];
  const int lbase = s * CHL;
  for (int l0 = 0; l0 < CHL; l0 += 16) {
    for (int i = threadIdx.x; i < 16 * Nz; i += blockDim.x) {
      const int ll = i >> 4, n = i & 15;
      Bs[ll][n] = Bm[((size_t)bg * Lz + lbase + l0 + ll) * Nz + n];
    }
    __syncthreads();
    if (l0 + 16 < CHL) {   // gfx1250 global_prefetch for the next sub-chunk
      __builtin_prefetch(&dt[((size_t)bg * Lz + lbase + l0 + 16) * Ez + e], 0, 1);
      __builtin_prefetch(&uc[((size_t)bg * Lz + lbase + l0 + 16) * Ez + e], 0, 1);
    }
    for (int ll = 0; ll < 16; ++ll) {
      const size_t ei = ((size_t)bg * Lz + lbase + l0 + ll) * Ez + e;
      const float dtv = dt[ei], ucv = uc[ei];
      const float duc = dtv * ucv;
      sdt += dtv;
#pragma unroll
      for (int n = 0; n < Nz; ++n)
        h[n] = __expf(dtv * A[n]) * h[n] + duc * Bs[ll][n];
    }
    __syncthreads();
  }
  const size_t base = ((size_t)bg * Ez + e) * Sz + s;
  sdt_buf[base] = sdt;
#pragma unroll
  for (int n = 0; n < Nz; ++n) q_buf[base * Nz + n] = h[n];
}

// ---------- K6b: sequential combine across 16 chunks (tiny) ----------
__global__ void k_scan_b(const float* __restrict__ sdt_buf, const float* __restrict__ q_buf,
                         const float* __restrict__ Alog, float* __restrict__ hin) {
  const int bg = blockIdx.x, g = bg & 3, e = threadIdx.x;
  float A[Nz];
#pragma unroll
  for (int n = 0; n < Nz; ++n) A[n] = -__expf(Alog[((size_t)g * Ez + e) * Nz + n]);
  float h[Nz];
#pragma unroll
  for (int n = 0; n < Nz; ++n) h[n] = 0.f;
  for (int s = 0; s < Sz; ++s) {
    const size_t base = ((size_t)bg * Ez + e) * Sz + s;
#pragma unroll
    for (int n = 0; n < Nz; ++n) hin[base * Nz + n] = h[n];
    const float sdt = sdt_buf[base];
#pragma unroll
    for (int n = 0; n < Nz; ++n)
      h[n] = __expf(A[n] * sdt) * h[n] + q_buf[base * Nz + n];
  }
}

// ---------- K6c: chunked scan pass C: re-run chunk from h_in, emit gated y (bf16) ----------
__global__ void k_scan_c(const float* __restrict__ xz, const float* __restrict__ uc,
                         const float* __restrict__ dt, const float* __restrict__ Bm,
                         const float* __restrict__ Cm, const float* __restrict__ Alog,
                         const float* __restrict__ Dskip, const float* __restrict__ hin,
                         u16* __restrict__ ybf) {
  const int bg = blockIdx.x >> 4;
  const int s  = blockIdx.x & 15;
  const int g = bg & 3, e = threadIdx.x;
  float A[Nz];
#pragma unroll
  for (int n = 0; n < Nz; ++n) A[n] = -__expf(Alog[((size_t)g * Ez + e) * Nz + n]);
  const float Dsk = Dskip[g * Ez + e];
  const size_t hbase = (((size_t)bg * Ez + e) * Sz + s) * Nz;
  float h[Nz];
#pragma unroll
  for (int n = 0; n < Nz; ++n) h[n] = hin[hbase + n];
  __shared__ float Bs[16][Nz], Cs[16][Nz];
  const int lbase = s * CHL;
  for (int l0 = 0; l0 < CHL; l0 += 16) {
    for (int i = threadIdx.x; i < 16 * Nz; i += blockDim.x) {
      const int ll = i >> 4, n = i & 15;
      const size_t bi = ((size_t)bg * Lz + lbase + l0 + ll) * Nz + n;
      Bs[ll][n] = Bm[bi]; Cs[ll][n] = Cm[bi];
    }
    __syncthreads();
    if (l0 + 16 < CHL) {
      __builtin_prefetch(&dt[((size_t)bg * Lz + lbase + l0 + 16) * Ez + e], 0, 1);
      __builtin_prefetch(&uc[((size_t)bg * Lz + lbase + l0 + 16) * Ez + e], 0, 1);
    }
    for (int ll = 0; ll < 16; ++ll) {
      const int l = lbase + l0 + ll;
      const size_t ei = ((size_t)bg * Lz + l) * Ez + e;
      const float dtv = dt[ei], ucv = uc[ei];
      const float duc = dtv * ucv;
      float y = 0.f;
#pragma unroll
      for (int n = 0; n < Nz; ++n) {
        h[n] = __expf(dtv * A[n]) * h[n] + duc * Bs[ll][n];
        y += h[n] * Cs[ll][n];
      }
      const float z = xz[((size_t)bg * Lz + l) * F2 + Ez + e];
      const float yv = (y + ucv * Dsk) * (z * sigm(z));
      ybf[ei] = f2bf(yv);
    }
    __syncthreads();
  }
}

// ---------- K7: out_proj GEMM + group residual + gate -> xwt bf16 ----------
__global__ void k_gemm_out(const u16* __restrict__ ybf, const u16* __restrict__ Wot,
                           const float* __restrict__ xn, const float* __restrict__ wgate,
                           u16* __restrict__ xwt) {
  const int wave = blockIdx.x * (blockDim.x >> 5) + (threadIdx.x >> 5);
  const int mt = wave & 127;
  const int bg = wave >> 7;
  const int b = bg >> 2, g = bg & 3;
  const u16* A  = ybf + ((size_t)bg * Lz + mt * 16) * Ez;
  const u16* Bp = Wot + (size_t)g * Ez * Dz;
  v8f acc[4];
#pragma unroll
  for (int i = 0; i < 4; ++i) acc[i] = vzero();
  wmma_16x64(A, Ez, Bp, Dz, Ez, acc);
  const int lane = threadIdx.x & 31, n = lane & 15, hi = lane >> 4;
#pragma unroll
  for (int nt = 0; nt < 4; ++nt)
#pragma unroll
    for (int r = 0; r < 8; ++r) {
      const int l = mt * 16 + hi * 8 + r;
      const int c = g * 64 + nt * 16 + n;
      float v = acc[nt][r] + xn[((size_t)b * Lz + l) * Cz + c];
      v *= wgate[b * Cz + c];
      xwt[((size_t)b * Lz + l) * Cz + c] = f2bf(v);
    }
}

// ---------- K8: final projection GEMM + bias + residual -> out (B,C,L) f32 ----------
__global__ void k_gemm_proj(const u16* __restrict__ xwt, const u16* __restrict__ Wp,
                            const float* __restrict__ proj_b, const float* __restrict__ x,
                            float* __restrict__ out) {
  const int wave = blockIdx.x * (blockDim.x >> 5) + (threadIdx.x >> 5);
  const int n64 = wave & 3;
  const int mt  = (wave >> 2) & 127;
  const int b   = wave >> 9;
  const u16* A  = xwt + ((size_t)b * Lz + mt * 16) * Cz;
  const u16* Bp = Wp + n64 * 64;
  v8f acc[4];
#pragma unroll
  for (int i = 0; i < 4; ++i) acc[i] = vzero();
  wmma_16x64(A, Cz, Bp, Cz, Cz, acc);
  const int lane = threadIdx.x & 31, n = lane & 15, hi = lane >> 4;
#pragma unroll
  for (int nt = 0; nt < 4; ++nt) {
    const int o = n64 * 64 + nt * 16 + n;
    const float pb = proj_b[o];
    const size_t base = ((size_t)(b * Cz + o)) * Lz + mt * 16 + hi * 8;
#pragma unroll
    for (int r = 0; r < 8; ++r) out[base + r] = acc[nt][r] + pb + x[base + r];
  }
}

// ---------- host ----------
extern "C" void kernel_launch(void* const* d_in, const int* in_sizes, int n_in,
                              void* d_out, int out_size, void* d_ws, size_t ws_size,
                              hipStream_t stream) {
  (void)in_sizes; (void)n_in; (void)out_size; (void)ws_size;
  const float* x       = (const float*)d_in[0];
  const float* ln_w    = (const float*)d_in[1];
  const float* ln_b    = (const float*)d_in[2];
  const float* cam_w1  = (const float*)d_in[3];
  const float* cam_b1  = (const float*)d_in[4];
  const float* cam_w2  = (const float*)d_in[5];
  const float* cam_b2  = (const float*)d_in[6];
  const float* proj_w  = (const float*)d_in[7];
  const float* proj_b  = (const float*)d_in[8];
  const float* g_ln_w  = (const float*)d_in[9];
  const float* g_ln_b  = (const float*)d_in[10];
  const float* g_in_w  = (const float*)d_in[11];
  const float* g_conv_w= (const float*)d_in[12];
  const float* g_conv_b= (const float*)d_in[13];
  const float* g_xproj = (const float*)d_in[14];
  const float* g_dt_w  = (const float*)d_in[15];
  const float* g_dt_b  = (const float*)d_in[16];
  const float* g_Alog  = (const float*)d_in[17];
  const float* g_Dskip = (const float*)d_in[18];
  const float* g_out_w = (const float*)d_in[19];
  float* out = (float*)d_out;

  char* w = (char*)d_ws;
  auto alloc = [&](size_t bytes) { char* p = w; w += (bytes + 255) & ~(size_t)255; return (void*)p; };
  float* xn     = (float*)alloc((size_t)Bz * Lz * Cz * 4);
  float* s_part = (float*)alloc((size_t)Bz * 64 * Cz * 4);
  float* wgate  = (float*)alloc((size_t)Bz * Cz * 4);
  u16*   hg     = (u16*)  alloc((size_t)Bz * Gz * Lz * Dz * 2);
  u16*   Wt     = (u16*)  alloc((size_t)Gz * Dz * F2 * 2);
  u16*   Wot    = (u16*)  alloc((size_t)Gz * Ez * Dz * 2);
  u16*   Wp     = (u16*)  alloc((size_t)Cz * Cz * 2);
  float* xz     = (float*)alloc((size_t)Bz * Gz * Lz * F2 * 4);
  float* uc     = (float*)alloc((size_t)Bz * Gz * Lz * Ez * 4);
  float* dtb    = (float*)alloc((size_t)Bz * Gz * Lz * Ez * 4);
  float* Bm     = (float*)alloc((size_t)Bz * Gz * Lz * Nz * 4);
  float* Cm     = (float*)alloc((size_t)Bz * Gz * Lz * Nz * 4);
  u16*   ybf    = (u16*)  alloc((size_t)Bz * Gz * Lz * Ez * 2);
  u16*   xwt    = (u16*)  alloc((size_t)Bz * Lz * Cz * 2);
  float* sdtb   = (float*)alloc((size_t)Bz * Gz * Ez * Sz * 4);
  float* qbuf   = (float*)alloc((size_t)Bz * Gz * Ez * Sz * Nz * 4);
  float* hin    = (float*)alloc((size_t)Bz * Gz * Ez * Sz * Nz * 4);

  k_ln<<<Bz * (Lz / 32), 256, 0, stream>>>(x, ln_w, ln_b, g_ln_w, g_ln_b, xn, hg, s_part);
  k_gate<<<Bz, 256, 0, stream>>>(s_part, cam_w1, cam_b1, cam_w2, cam_b2, wgate);
  k_convert<<<(Gz*Dz*F2 + Gz*Ez*Dz + Cz*Cz) / 256, 256, 0, stream>>>(g_in_w, g_out_w, proj_w, Wt, Wot, Wp);
  k_gemm_in<<<(Bz*Gz*(Lz/16)*4) / 8, 256, 0, stream>>>(hg, Wt, xz);
  k_conv<<<Bz * Gz * (Lz / 32), 256, 0, stream>>>(xz, g_conv_w, g_conv_b, g_xproj, g_dt_w, g_dt_b,
                                                  uc, dtb, Bm, Cm);
  k_scan_a<<<Bz * Gz * Sz, Ez, 0, stream>>>(dtb, uc, Bm, g_Alog, sdtb, qbuf);
  k_scan_b<<<Bz * Gz, Ez, 0, stream>>>(sdtb, qbuf, g_Alog, hin);
  k_scan_c<<<Bz * Gz * Sz, Ez, 0, stream>>>(xz, uc, dtb, Bm, Cm, g_Alog, g_Dskip, hin, ybf);
  k_gemm_out<<<(Bz*Gz*(Lz/16)) / 8, 256, 0, stream>>>(ybf, Wot, xn, wgate, xwt);
  k_gemm_proj<<<(Bz*(Lz/16)*4) / 8, 256, 0, stream>>>(xwt, Wp, proj_b, x, out);
}
